// MultiHeadAttention_39024072852132
// MI455X (gfx1250) — matrix operand order repaired and checked by reading, not compiled, not run on hardware
//
#include <hip/hip_runtime.h>
#include <math.h>

// ---------------- problem constants ----------------
#define BATCH 2
#define TQ 2048
#define TKK 2048
#define DMODEL 2048
#define NHEAD 16
#define DHEAD 128

typedef __bf16 bf16;
typedef __attribute__((ext_vector_type(16))) __bf16 v16bf;
typedef __attribute__((ext_vector_type(8)))  float  v8f;
typedef __attribute__((ext_vector_type(4)))  unsigned int u32x4;

#define WMMA_BF16(a, b, c) \
  __builtin_amdgcn_wmma_f32_16x16x32_bf16(false, (a), false, (b), (short)0, (c), false, false)

// ---- CDNA5 async global->LDS (16B per lane), tracked by ASYNCcnt ----
__device__ __forceinline__ void async_ld_b128(const void* gptr, const void* lptr) {
  unsigned lds = (unsigned)(uintptr_t)lptr;  // low 32 bits of generic = LDS offset
  unsigned long long ga = (unsigned long long)(uintptr_t)gptr;
  asm volatile("global_load_async_to_lds_b128 %0, %1, off"
               :: "v"(lds), "v"(ga) : "memory");
}
__device__ __forceinline__ void wait_async0() {
  asm volatile("s_wait_asynccnt 0" ::: "memory");
}

// ---- fragment loaders from LDS (pitch in bf16 elements, rows 16B-aligned) ----
// A 16x32 (MxK): lane&15 = row M; lane holds 16 contiguous K {0..15} or {16..31}.
__device__ __forceinline__ v16bf ld_frag_a(const bf16* p, int pitch, int lane) {
  const bf16* q = p + (lane & 15) * pitch + ((lane >> 4) << 4);
  union { v16bf v; u32x4 u[2]; } r;
  r.u[0] = *(const u32x4*)(q);
  r.u[1] = *(const u32x4*)(q + 8);
  return r.v;
}
// B 32x16 (KxN), source stored [N][K] row-major: lane&15 = col N; 16 contiguous K.
__device__ __forceinline__ v16bf ld_frag_b(const bf16* p, int pitch, int lane) {
  const bf16* q = p + (lane & 15) * pitch + ((lane >> 4) << 4);
  union { v16bf v; u32x4 u[2]; } r;
  r.u[0] = *(const u32x4*)(q);
  r.u[1] = *(const u32x4*)(q + 8);
  return r.v;
}

// ---------------- f32 -> bf16 conversion (8 elems/thread, vector ld/st) ----------------
__global__ __launch_bounds__(256) void cvt_f32_bf16(const float* __restrict__ in,
                                                    bf16* __restrict__ out, int n) {
  int i = (blockIdx.x * 256 + threadIdx.x) * 8;
  if (i >= n) return;
  union { u32x4 u; float f[4]; } a, b;
  union { u32x4 u; bf16 h[8]; } o;
  a.u = *(const u32x4*)(in + i);
  b.u = *(const u32x4*)(in + i + 4);
#pragma unroll
  for (int j = 0; j < 4; ++j) { o.h[j] = (bf16)a.f[j]; o.h[4 + j] = (bf16)b.f[j]; }
  *(u32x4*)(out + i) = o.u;
}

// ---------------- GEMM: C[M,N] = A[M,K] @ W[N,K]^T ----------------
// Block tile 128(M) x 256(N), K step 32, double-buffered async LDS staging.
// 8 waves arranged 2(M) x 4(N); wave tile 64x64 = 4x4 WMMA accumulators.
#define BM 128
#define BN 256
#define BK 32
#define LP (BK + 8)   // 40 bf16 = 80B rows, 16B aligned

template <bool F32OUT>
__global__ __launch_bounds__(256) void gemm_bf16nt(const bf16* __restrict__ A,
                                                   const bf16* __restrict__ W,
                                                   bf16* __restrict__ Cb,
                                                   float* __restrict__ Cf,
                                                   int M, int N, int K) {
  __shared__ __align__(16) bf16 As[2][BM][LP];  // 20 KB
  __shared__ __align__(16) bf16 Ws[2][BN][LP];  // 40 KB

  const int tid  = threadIdx.x;
  const int lane = tid & 31;
  const int wave = tid >> 5;
  const int wm   = wave & 1;    // 64-row M slab
  const int wn   = wave >> 1;   // 64-col N slab
  const int m0   = blockIdx.y * BM;
  const int n0   = blockIdx.x * BN;

  v8f acc[4][4] = {};

  auto stage = [&](int buf, int k0) {
#pragma unroll
    for (int c = 0; c < 2; ++c) {   // A: 128x32 = 4096 elems
      int e = (tid + 256 * c) * 8;
      int r = e >> 5, col = e & 31;
      async_ld_b128(&A[(size_t)(m0 + r) * K + k0 + col], &As[buf][r][col]);
    }
#pragma unroll
    for (int c = 0; c < 4; ++c) {   // W: 256x32 = 8192 elems
      int e = (tid + 256 * c) * 8;
      int r = e >> 5, col = e & 31;
      async_ld_b128(&W[(size_t)(n0 + r) * K + k0 + col], &Ws[buf][r][col]);
    }
  };

  stage(0, 0);
  wait_async0();
  __syncthreads();

  int cur = 0;
  for (int k0 = 0; k0 < K; k0 += BK) {
    const int nxt = cur ^ 1;
    const bool more = (k0 + BK) < K;
    if (more) stage(nxt, k0 + BK);  // async loads overlap the WMMAs below

    // preload all fragments, single dscnt wait, then 16 back-to-back WMMAs
    v16bf af[4], bw[4];
#pragma unroll
    for (int mi = 0; mi < 4; ++mi)
      af[mi] = ld_frag_a(&As[cur][64 * wm + 16 * mi][0], LP, lane);
#pragma unroll
    for (int nt = 0; nt < 4; ++nt)
      bw[nt] = ld_frag_b(&Ws[cur][64 * wn + 16 * nt][0], LP, lane);
#pragma unroll
    for (int nt = 0; nt < 4; ++nt)
#pragma unroll
      for (int mi = 0; mi < 4; ++mi)
        acc[mi][nt] = WMMA_BF16(af[mi], bw[nt], acc[mi][nt]);

    if (more) wait_async0();
    __syncthreads();
    cur = nxt;
  }

  // C layout: VGPR j -> rows {j, j+8}; lane halves select.
  const int hi = lane >> 4, nn = lane & 15;
#pragma unroll
  for (int mi = 0; mi < 4; ++mi)
#pragma unroll
    for (int nt = 0; nt < 4; ++nt) {
      const int rb = m0 + 64 * wm + 16 * mi + 8 * hi;
      const int cc = n0 + 64 * wn + 16 * nt + nn;
      size_t idx = (size_t)rb * N + cc;
#pragma unroll
      for (int j = 0; j < 8; ++j) {
        if constexpr (F32OUT) Cf[idx] = acc[mi][nt][j];
        else                  Cb[idx] = (bf16)acc[mi][nt][j];
        idx += (size_t)N;
      }
    }
}

// ---------------- flash attention: one (b, h, 128-q-tile) per workgroup ----------------
#define QT 128
#define KC 64
#define QP 136   // 272B rows
#define VP 72    // 144B rows
#define SP 68    // float pitch, 272B rows

__global__ __launch_bounds__(256) void attn_flash(const bf16* __restrict__ Qg,
                                                  const bf16* __restrict__ Kg,
                                                  const bf16* __restrict__ Vg,
                                                  const unsigned char* __restrict__ am,
                                                  const unsigned char* __restrict__ kpm,
                                                  bf16* __restrict__ AO) {
  __shared__ __align__(16) bf16  Qs[QT][QP];
  __shared__ __align__(16) bf16  Ks[KC][QP];
  __shared__ __align__(16) bf16  Vt[DHEAD][VP];   // transposed: Vt[d][k]
  __shared__ __align__(16) float Sf[QT][SP];
  __shared__ __align__(16) bf16  Pb[QT][VP];
  __shared__ float m_s[QT], l_s[QT], alpha_s[QT];

  const int tid  = threadIdx.x;
  const int lane = tid & 31;
  const int wave = tid >> 5;
  const int hi   = lane >> 4, nn = lane & 15;
  const int q0   = blockIdx.x * QT;
  const int h    = blockIdx.y;
  const int b    = blockIdx.z;
  const float rsc = 0.08838834764831845f;  // 1/sqrt(128)

  // stage Q tile (async)
#pragma unroll
  for (int c = 0; c < 8; ++c) {
    int e = (tid + 256 * c) * 8;
    int r = e >> 7, d = e & 127;
    async_ld_b128(&Qg[((size_t)(b * TQ + q0 + r)) * DMODEL + h * DHEAD + d],
                  &Qs[r][d]);
  }
  if (tid < QT) { m_s[tid] = -INFINITY; l_s[tid] = 0.0f; }

  v8f Oacc[8] = {};

  for (int kc = 0; kc < TKK; kc += KC) {
    // stage K chunk (async) and transposed V chunk (via regs)
#pragma unroll
    for (int c = 0; c < 4; ++c) {
      int e = (tid + 256 * c) * 8;
      int r = e >> 7, d = e & 127;
      async_ld_b128(&Kg[((size_t)(b * TKK + kc + r)) * DMODEL + h * DHEAD + d],
                    &Ks[r][d]);
      union { u32x4 v; bf16 x[8]; } t;
      t.v = *(const u32x4*)&Vg[((size_t)(b * TKK + kc + r)) * DMODEL + h * DHEAD + d];
#pragma unroll
      for (int j = 0; j < 8; ++j) Vt[d + j][r] = t.x[j];
    }
    wait_async0();
    __syncthreads();

    // S = Q K^T (per wave: 16 q-rows x 64 k-cols)
    v8f sacc[4] = {};
#pragma unroll
    for (int dk = 0; dk < DHEAD; dk += 32) {
      v16bf aq = ld_frag_a(&Qs[16 * wave][dk], QP, lane);
#pragma unroll
      for (int nt = 0; nt < 4; ++nt) {
        v16bf bk = ld_frag_b(&Ks[16 * nt][dk], QP, lane);
        sacc[nt] = WMMA_BF16(aq, bk, sacc[nt]);
      }
    }
#pragma unroll
    for (int nt = 0; nt < 4; ++nt)
#pragma unroll
      for (int j = 0; j < 8; ++j)
        Sf[16 * wave + j + 8 * hi][16 * nt + nn] = sacc[nt][j] * rsc;
    __syncthreads();

    // online softmax: thread r < 128 owns q-row r
    if (tid < QT) {
      const int qg = q0 + tid;
      unsigned long long bits = 0ull;
      float rmax = -INFINITY;
#pragma unroll 8
      for (int c = 0; c < KC; ++c) {
        bool msk = am[(size_t)qg * TKK + kc + c] || kpm[(size_t)b * TKK + kc + c];
        if (msk) bits |= (1ull << c);
        else     rmax = fmaxf(rmax, Sf[tid][c]);
      }
      float mold = m_s[tid];
      float mnew = fmaxf(mold, rmax);
      float alpha, lsum = 0.0f;
      if (mnew == -INFINITY) {
        alpha = 1.0f;
#pragma unroll 8
        for (int c = 0; c < KC; ++c) Pb[tid][c] = (bf16)0.0f;
      } else {
        alpha = (mold == -INFINITY) ? 0.0f : __expf(mold - mnew);
#pragma unroll 8
        for (int c = 0; c < KC; ++c) {
          float p = ((bits >> c) & 1) ? 0.0f : __expf(Sf[tid][c] - mnew);
          Pb[tid][c] = (bf16)p;
          lsum += p;
        }
      }
      m_s[tid] = mnew;
      l_s[tid] = l_s[tid] * alpha + lsum;
      alpha_s[tid] = alpha;
    }
    __syncthreads();

    // rescale O and accumulate P @ V
    float al[8];
#pragma unroll
    for (int j = 0; j < 8; ++j) al[j] = alpha_s[16 * wave + j + 8 * hi];
#pragma unroll
    for (int dt = 0; dt < 8; ++dt)
#pragma unroll
      for (int j = 0; j < 8; ++j) Oacc[dt][j] *= al[j];
#pragma unroll
    for (int kk = 0; kk < KC; kk += 32) {
      v16bf ap = ld_frag_a(&Pb[16 * wave][kk], VP, lane);
#pragma unroll
      for (int dt = 0; dt < 8; ++dt) {
        v16bf bv = ld_frag_b(&Vt[16 * dt][kk], VP, lane);
        Oacc[dt] = WMMA_BF16(ap, bv, Oacc[dt]);
      }
    }
    __syncthreads();
  }

  // normalize and write attention output (bf16, [B,TQ,D] col = h*128+d)
  float linv[8];
#pragma unroll
  for (int j = 0; j < 8; ++j) linv[j] = 1.0f / l_s[16 * wave + j + 8 * hi];
#pragma unroll
  for (int dt = 0; dt < 8; ++dt)
#pragma unroll
    for (int j = 0; j < 8; ++j) {
      int rit = 16 * wave + j + 8 * hi;
      AO[((size_t)(b * TQ + q0 + rit)) * DMODEL + h * DHEAD + 16 * dt + nn] =
          (bf16)(Oacc[dt][j] * linv[j]);
    }
}

// ---------------- host-side orchestration ----------------
extern "C" void kernel_launch(void* const* d_in, const int* in_sizes, int n_in,
                              void* d_out, int out_size, void* d_ws, size_t ws_size,
                              hipStream_t stream) {
  const float* q  = (const float*)d_in[0];
  const float* k  = (const float*)d_in[1];
  const float* v  = (const float*)d_in[2];
  const float* Wq = (const float*)d_in[3];
  const float* Wk = (const float*)d_in[4];
  const float* Wv = (const float*)d_in[5];
  const float* Wo = (const float*)d_in[6];
  const unsigned char* am  = (const unsigned char*)d_in[7];
  const unsigned char* kpm = (const unsigned char*)d_in[8];
  float* out = (float*)d_out;

  const int NT = BATCH * TQ * DMODEL;    // 8,388,608
  const int DD = DMODEL * DMODEL;        // 4,194,304
  const int M  = BATCH * TQ;             // 4096

  size_t off = 0;
  auto carve = [&](size_t bytes) {
    void* p = (char*)d_ws + off;
    off += (bytes + 255) & ~(size_t)255;
    return p;
  };
  bf16* qb  = (bf16*)carve((size_t)NT * 2);
  bf16* kb  = (bf16*)carve((size_t)NT * 2);
  bf16* vb  = (bf16*)carve((size_t)NT * 2);
  bf16* Wqb = (bf16*)carve((size_t)DD * 2);
  bf16* Wkb = (bf16*)carve((size_t)DD * 2);
  bf16* Wvb = (bf16*)carve((size_t)DD * 2);
  bf16* Wob = (bf16*)carve((size_t)DD * 2);
  bf16* Qp  = (bf16*)carve((size_t)NT * 2);
  bf16* Kp  = (bf16*)carve((size_t)NT * 2);
  bf16* Vp  = (bf16*)carve((size_t)NT * 2);
  bf16* AO  = (bf16*)carve((size_t)NT * 2);

  cvt_f32_bf16<<<NT / (256 * 8), 256, 0, stream>>>(q,  qb,  NT);
  cvt_f32_bf16<<<NT / (256 * 8), 256, 0, stream>>>(k,  kb,  NT);
  cvt_f32_bf16<<<NT / (256 * 8), 256, 0, stream>>>(v,  vb,  NT);
  cvt_f32_bf16<<<DD / (256 * 8), 256, 0, stream>>>(Wq, Wqb, DD);
  cvt_f32_bf16<<<DD / (256 * 8), 256, 0, stream>>>(Wk, Wkb, DD);
  cvt_f32_bf16<<<DD / (256 * 8), 256, 0, stream>>>(Wv, Wvb, DD);
  cvt_f32_bf16<<<DD / (256 * 8), 256, 0, stream>>>(Wo, Wob, DD);

  dim3 gg(DMODEL / BN, M / BM);  // (8, 32)
  gemm_bf16nt<false><<<gg, 256, 0, stream>>>(qb, Wqb, Qp, nullptr, M, DMODEL, DMODEL);
  gemm_bf16nt<false><<<gg, 256, 0, stream>>>(kb, Wkb, Kp, nullptr, M, DMODEL, DMODEL);
  gemm_bf16nt<false><<<gg, 256, 0, stream>>>(vb, Wvb, Vp, nullptr, M, DMODEL, DMODEL);

  dim3 ga(TQ / QT, NHEAD, BATCH);  // (16, 16, 2)
  attn_flash<<<ga, 256, 0, stream>>>(Qp, Kp, Vp, am, kpm, AO);

  gemm_bf16nt<true><<<gg, 256, 0, stream>>>(AO, Wob, nullptr, out, M, DMODEL, DMODEL);
}
